// GraphMultiHeadAttention_25082609008821
// MI455X (gfx1250) — compile-verified, matrix operand order checked by reading, not run on hardware
//
#include <hip/hip_runtime.h>
#include <hip/hip_bf16.h>

// ---------------- problem constants ----------------
#define PB   4
#define PS   512
#define PD   1024
#define PH   16
#define PDh  64
#define PT   32
#define PE   16
#define PSCALE 0.125f   // Dh^-0.5 = 64^-0.5

typedef __attribute__((ext_vector_type(16))) __bf16 v16bf;
typedef __attribute__((ext_vector_type(8)))  float  v8f;

struct Frag32 { uint4 lo, hi; };   // 32 bytes == 16 bf16

// ---------------- helpers ----------------
__device__ __forceinline__ unsigned short f2bf(float f) {
    unsigned u = __builtin_bit_cast(unsigned, f);
    u += 0x7FFFu + ((u >> 16) & 1u);      // round-to-nearest-even
    return (unsigned short)(u >> 16);
}
__device__ __forceinline__ float bf2f(unsigned short b) {
    return __builtin_bit_cast(float, (unsigned)b << 16);
}

__device__ __forceinline__ v8f wmma_bf16(v16bf a, v16bf b, v8f c) {
    // (neg_a, A, neg_b, B, c_mod, C, reuse_a, reuse_b)
    return __builtin_amdgcn_wmma_f32_16x16x32_bf16(false, a, false, b, (short)0, c, false, false);
}

// CDNA5 async global->LDS (VGLOBAL opcode 98): LDS dest addr in VDST VGPR,
// global addr in VADDR pair; tracked by ASYNCcnt.
__device__ __forceinline__ void async_ld_b128(unsigned lds_off, const void* g) {
    asm volatile("global_load_async_to_lds_b128 %0, %1, off"
                 :: "v"(lds_off), "v"((unsigned long long)(size_t)g)
                 : "memory");
}
__device__ __forceinline__ void wait_async0() {
    asm volatile("s_wait_asynccnt 0x0" ::: "memory");
}
// generic pointer into LDS: low 32 bits are the wave-relative LDS byte address
__device__ __forceinline__ unsigned lds_addr_of(const void* p) {
    return (unsigned)(size_t)p;
}

// A/B fragment, 16 rows x 32 K (bf16), source row-major with K contiguous.
// ISA layout: lanes 0-15 (row = lane)  hold K 0..7 then 16..23;
//             lanes 16-31 (row=lane-16) hold K 8..15 then 24..31.
__device__ __forceinline__ v16bf load_frag_g(const unsigned short* base, int row0,
                                             int stride, int k0, int lane) {
    int r = row0 + (lane & 15);
    int k = k0 + ((lane & 16) ? 8 : 0);
    const unsigned short* p = base + (size_t)r * stride + k;
    Frag32 fr;
    fr.lo = *(const uint4*)p;          // K .. K+7
    fr.hi = *(const uint4*)(p + 16);   // K+16 .. K+23
    return __builtin_bit_cast(v16bf, fr);
}

// same, from LDS tile stored as 16 uints (32 bf16) per row
__device__ __forceinline__ v16bf load_frag_lds(const unsigned* base, int row0, int lane) {
    const unsigned* p = base + (row0 + (lane & 15)) * 16 + ((lane & 16) ? 4 : 0);
    Frag32 fr;
    fr.lo = *(const uint4*)p;
    fr.hi = *(const uint4*)(p + 8);
    return __builtin_bit_cast(v16bf, fr);
}

// probability buffer in LDS: 32 rows x 256 uints (512 bf16 keys) per row
__device__ __forceinline__ v16bf load_frag_p(const unsigned* pbuf, int row0, int k0, int lane) {
    const unsigned* p = pbuf + (row0 + (lane & 15)) * 256 + (k0 >> 1) + ((lane & 16) ? 4 : 0);
    Frag32 fr;
    fr.lo = *(const uint4*)p;
    fr.hi = *(const uint4*)(p + 8);
    return __builtin_bit_cast(v16bf, fr);
}

// ---------------- kernel 1: fp32 -> bf16 ----------------
__global__ void k_cvt_bf16(const float* __restrict__ in, unsigned short* __restrict__ out, int n) {
    int i = (blockIdx.x * blockDim.x + threadIdx.x) * 2;
    if (i + 1 < n) {
        unsigned u = (unsigned)f2bf(in[i]) | ((unsigned)f2bf(in[i + 1]) << 16);
        *(unsigned*)(out + i) = u;
    }
}

// ---------------- kernel 2: bf16 WMMA GEMM  C = A(MxK) * B(NxK)^T + bias -----
// Block tile 128x64, K-step 32, double-buffered LDS filled by async loads.
// Each wave owns a 2x2 grid of 16x16 C tiles -> 4 WMMAs per K-step.
// mode 0: store bf16 head-major        dst[((b*H+h)*S + s)*64 + dh]
// mode 1: store f32 row-major          dst[gm*N + gn]
// mode 2: store bf16 head-major, TRANSPOSED per head: dst[((b*H+h)*64 + dh)*S + s]
__global__ __launch_bounds__(256) void k_gemm_bf16(
    const unsigned short* __restrict__ A, const unsigned short* __restrict__ Bw,
    const float* __restrict__ bias, float* __restrict__ outF,
    unsigned short* __restrict__ outBF, int M, int N, int K, int mode)
{
    __shared__ __align__(16) unsigned As[2][128 * 16];   // 2 x (128 rows x 32 bf16)
    __shared__ __align__(16) unsigned Bs[2][64 * 16];    // 2 x (64 rows x 32 bf16)

    const int tid = threadIdx.x, lane = tid & 31, wave = tid >> 5;
    const int m0 = blockIdx.x * 128, n0 = blockIdx.y * 64;
    const int mtb = (wave & 3) * 2, ntb = (wave >> 2) * 2;

    v8f acc[2][2] = {};

    const unsigned asbase = lds_addr_of(&As[0][0]);
    const unsigned bsbase = lds_addr_of(&Bs[0][0]);
    const int row = tid >> 2, seg = tid & 3;  // 64 B-rows, 2x64 A-rows per thread pass

    auto issue = [&](int buf, int k0) {
        const unsigned aoff = asbase + (unsigned)buf * (128 * 16 * 4);
        const unsigned boff = bsbase + (unsigned)buf * (64 * 16 * 4);
        async_ld_b128(aoff + (unsigned)((row       * 16 + seg * 4) * 4),
                      A + (size_t)(m0 + row)      * K + k0 + seg * 8);
        async_ld_b128(aoff + (unsigned)(((row + 64) * 16 + seg * 4) * 4),
                      A + (size_t)(m0 + row + 64) * K + k0 + seg * 8);
        async_ld_b128(boff + (unsigned)((row       * 16 + seg * 4) * 4),
                      Bw + (size_t)(n0 + row)     * K + k0 + seg * 8);
    };

    issue(0, 0);
    int buf = 0;
    for (int k0 = 0; k0 < K; k0 += 32, buf ^= 1) {
        wait_async0();          // our async stores into LDS landed
        __syncthreads();        // everyone's landed; prev compute drained
        if (k0 + 32 < K) issue(buf ^ 1, k0 + 32);
        const unsigned* Ab = &As[buf][0];
        const unsigned* Bb = &Bs[buf][0];
        v16bf a0 = load_frag_lds(Ab, (mtb + 0) * 16, lane);
        v16bf a1 = load_frag_lds(Ab, (mtb + 1) * 16, lane);
        v16bf b0 = load_frag_lds(Bb, (ntb + 0) * 16, lane);
        v16bf b1 = load_frag_lds(Bb, (ntb + 1) * 16, lane);
        acc[0][0] = wmma_bf16(a0, b0, acc[0][0]);
        acc[0][1] = wmma_bf16(a0, b1, acc[0][1]);
        acc[1][0] = wmma_bf16(a1, b0, acc[1][0]);
        acc[1][1] = wmma_bf16(a1, b1, acc[1][1]);
    }

    // C layout: lanes 0-15 N=lane, M=vgpr; lanes 16-31 N=lane-16, M=vgpr+8
    #pragma unroll
    for (int i = 0; i < 2; ++i) {
        const int gmb = m0 + (mtb + i) * 16 + ((lane & 16) ? 8 : 0);
        #pragma unroll
        for (int j = 0; j < 2; ++j) {
            const int gn = n0 + (ntb + j) * 16 + (lane & 15);
            #pragma unroll
            for (int r = 0; r < 8; ++r) {
                int gm = gmb + r;
                float v = acc[i][j][r] + bias[gn];
                if (mode == 1) {
                    outF[(size_t)gm * N + gn] = v;
                } else {
                    int bb = gm / PS, s = gm % PS, hh = gn >> 6, dh = gn & 63;
                    size_t idx = (mode == 0)
                        ? ((size_t)(bb * PH + hh) * PS + s) * 64 + dh
                        : ((size_t)(bb * PH + hh) * 64 + dh) * PS + s;
                    outBF[idx] = f2bf(v);
                }
            }
        }
    }
}

// ---------------- kernel 3: hop/edge bias tables ----------------
__global__ __launch_bounds__(64) void k_tables(
    const unsigned short* __restrict__ qh_bf, const unsigned short* __restrict__ kh_bf,
    const float* __restrict__ q_hop_emb, const float* __restrict__ q_edge_emb,
    const float* __restrict__ k_hop_emb, const float* __restrict__ k_edge_emb,
    float* __restrict__ qhopT, float* __restrict__ qedgeT,
    float* __restrict__ khopT, float* __restrict__ kedgeT)
{
    __shared__ float qrow[64], krow[64];
    const int gid = blockIdx.x;              // (b*H+h)*S + s
    const int tid = threadIdx.x;
    const size_t base = (size_t)gid * 64;
    qrow[tid] = bf2f(qh_bf[base + tid]);
    krow[tid] = bf2f(kh_bf[base + tid]);
    __syncthreads();
    const int h = (gid >> 9) & (PH - 1);
    if (tid < PT) {
        const float* qt = q_hop_emb + (size_t)tid * PD + h * 64;
        const float* kt = k_hop_emb + (size_t)tid * PD + h * 64;
        float sq = 0.f, sk = 0.f;
        #pragma unroll 8
        for (int d = 0; d < 64; ++d) { sq += qrow[d] * qt[d]; sk += krow[d] * kt[d]; }
        qhopT[(size_t)gid * PT + tid] = sq;
        khopT[(size_t)gid * PT + tid] = sk;
    } else if (tid < PT + PE) {
        int e = tid - PT;
        const float* qt = q_edge_emb + (size_t)e * PD + h * 64;
        const float* kt = k_edge_emb + (size_t)e * PD + h * 64;
        float sq = 0.f, sk = 0.f;
        #pragma unroll 8
        for (int d = 0; d < 64; ++d) { sq += qrow[d] * qt[d]; sk += krow[d] * kt[d]; }
        qedgeT[(size_t)gid * PE + e] = sq;
        kedgeT[(size_t)gid * PE + e] = sk;
    }
}

// ---------------- kernel 4: fused attention (per (b,h) x 32-row q tile) -----
__global__ __launch_bounds__(256) void k_attn(
    const unsigned short* __restrict__ qh_bf, const unsigned short* __restrict__ kh_bf,
    const unsigned short* __restrict__ vT_bf,
    const float* __restrict__ qhopT, const float* __restrict__ qedgeT,
    const float* __restrict__ khopT, const float* __restrict__ kedgeT,
    const int* __restrict__ distance, const int* __restrict__ edge_attr,
    const float* __restrict__ v_hop_emb, const float* __restrict__ v_edge_emb,
    unsigned short* __restrict__ attout_bf)
{
    __shared__ __align__(16) float sc[32 * 512];     // scores, later aliased as bf16 probs
    __shared__ float rowmax[32], rowsum[32], red[32][8];
    __shared__ float hop_acc[32][PT];
    __shared__ float edge_acc[32][PE];
    __shared__ float hoptab[PT][64];
    __shared__ float edgetab[PE][64];

    const int tid = threadIdx.x, lane = tid & 31, wave = tid >> 5;
    const int b = blockIdx.z, h = blockIdx.y, s0 = blockIdx.x * 32;
    const int bh = b * PH + h;

    for (int i = tid; i < PT * 64; i += 256) hoptab[i >> 6][i & 63]  = v_hop_emb[(size_t)(i >> 6) * PD + h * 64 + (i & 63)];
    for (int i = tid; i < PE * 64; i += 256) edgetab[i >> 6][i & 63] = v_edge_emb[(size_t)(i >> 6) * PD + h * 64 + (i & 63)];
    for (int i = tid; i < 32 * PT; i += 256) (&hop_acc[0][0])[i]  = 0.f;
    for (int i = tid; i < 32 * PE; i += 256) (&edge_acc[0][0])[i] = 0.f;

    const unsigned short* qbase = qh_bf + (size_t)bh * PS * 64;
    const unsigned short* kbase = kh_bf + (size_t)bh * PS * 64;
    const unsigned short* vbase = vT_bf + (size_t)bh * 64 * PS;
    const float* qhopB  = qhopT  + (size_t)bh * PS * PT;
    const float* khopB  = khopT  + (size_t)bh * PS * PT;
    const float* qedgeB = qedgeT + (size_t)bh * PS * PE;
    const float* kedgeB = kedgeT + (size_t)bh * PS * PE;
    const int* distB = distance  + (size_t)b * PS * PS;
    const int* edgeB = edge_attr + (size_t)b * PS * PS;

    // Q fragments (2 M-tiles x 2 K-windows), reused across all 32 key tiles
    v16bf aq[2][2];
    #pragma unroll
    for (int mt = 0; mt < 2; ++mt)
        #pragma unroll
        for (int kf = 0; kf < 2; ++kf)
            aq[mt][kf] = load_frag_g(qbase, s0 + mt * 16, 64, kf * 32, lane);

    // ---- scores = Q K^T + gathered biases, scaled ----
    for (int i = 0; i < 4; ++i) {
        const int nt = wave + i * 8;
        const int key0 = nt * 16;
        v16bf bk0 = load_frag_g(kbase, key0, 64, 0,  lane);
        v16bf bk1 = load_frag_g(kbase, key0, 64, 32, lane);
        #pragma unroll
        for (int mt = 0; mt < 2; ++mt) {
            v8f c = {};
            c = wmma_bf16(aq[mt][0], bk0, c);
            c = wmma_bf16(aq[mt][1], bk1, c);
            const int ncol  = key0 + (lane & 15);
            const int mbase = mt * 16 + ((lane & 16) ? 8 : 0);
            #pragma unroll
            for (int r = 0; r < 8; ++r) {
                const int m = mbase + r;
                const int qrow = s0 + m;
                const int d = distB[(size_t)qrow * PS + ncol];
                const int e = edgeB[(size_t)qrow * PS + ncol];
                float biasv = qhopB[(size_t)qrow * PT + d] + khopB[(size_t)ncol * PT + d]
                            + qedgeB[(size_t)qrow * PE + e] + kedgeB[(size_t)ncol * PE + e];
                sc[m * 512 + ncol] = (c[r] + biasv) * PSCALE;
            }
        }
    }
    __syncthreads();

    // ---- softmax stats: 8 threads per row ----
    const int r = tid >> 3, sub = tid & 7;
    float mx = -1e30f;
    for (int j = sub * 64; j < sub * 64 + 64; ++j) mx = fmaxf(mx, sc[r * 512 + j]);
    red[r][sub] = mx;
    __syncthreads();
    if (sub == 0) {
        float m2 = red[r][0];
        #pragma unroll
        for (int j = 1; j < 8; ++j) m2 = fmaxf(m2, red[r][j]);
        rowmax[r] = m2;
    }
    __syncthreads();
    const float rm = rowmax[r];
    float sm = 0.f;
    for (int j = sub * 64; j < sub * 64 + 64; ++j) sm += __expf(sc[r * 512 + j] - rm);
    red[r][sub] = sm;
    __syncthreads();
    if (sub == 0) {
        float s2 = 0.f;
        #pragma unroll
        for (int j = 0; j < 8; ++j) s2 += red[r][j];
        rowsum[r] = 1.0f / s2;
    }
    __syncthreads();

    // ---- pass 2: normalize, scatter into hop/edge buckets, repack as bf16 ----
    unsigned packed[32];
    #pragma unroll 4
    for (int i = 0; i < 32; ++i) {
        const int pidx = tid * 32 + i;
        const int rr = pidx >> 8, j = pidx & 255;
        const int k0 = 2 * j;
        const float f0 = sc[rr * 512 + k0], f1 = sc[rr * 512 + k0 + 1];
        const float inv = rowsum[rr], m0 = rowmax[rr];
        const float p0 = __expf(f0 - m0) * inv;
        const float p1 = __expf(f1 - m0) * inv;
        const int qrow = s0 + rr;
        const size_t qb = (size_t)qrow * PS;
        atomicAdd(&hop_acc[rr][distB[qb + k0]],      p0);
        atomicAdd(&hop_acc[rr][distB[qb + k0 + 1]],  p1);
        atomicAdd(&edge_acc[rr][edgeB[qb + k0]],     p0);
        atomicAdd(&edge_acc[rr][edgeB[qb + k0 + 1]], p1);
        packed[i] = (unsigned)f2bf(p0) | ((unsigned)f2bf(p1) << 16);
    }
    __syncthreads();
    unsigned* pbuf = (unsigned*)sc;
    #pragma unroll 4
    for (int i = 0; i < 32; ++i) pbuf[tid * 32 + i] = packed[i];
    __syncthreads();

    // ---- out = att @ V  (+ hop/edge table contributions) ----
    const int mt = wave & 1, nt = wave >> 1;
    v8f acc = {};
    for (int kw = 0; kw < 16; ++kw) {
        v16bf a  = load_frag_p(pbuf, mt * 16, kw * 32, lane);
        v16bf bv = load_frag_g(vbase, nt * 16, PS, kw * 32, lane);
        acc = wmma_bf16(a, bv, acc);
    }
    const int ncol  = nt * 16 + (lane & 15);
    const int mbase = mt * 16 + ((lane & 16) ? 8 : 0);
    #pragma unroll
    for (int rr = 0; rr < 8; ++rr) {
        const int m = mbase + rr;
        float v = acc[rr];
        #pragma unroll 8
        for (int t = 0; t < PT; ++t) v += hop_acc[m][t] * hoptab[t][ncol];
        #pragma unroll 8
        for (int e = 0; e < PE; ++e) v += edge_acc[m][e] * edgetab[e][ncol];
        const size_t dst = ((size_t)(b * PS + s0 + m)) * PD + h * 64 + ncol;
        attout_bf[dst] = f2bf(v);
    }
}

// ---------------- host launcher ----------------
extern "C" void kernel_launch(void* const* d_in, const int* in_sizes, int n_in,
                              void* d_out, int out_size, void* d_ws, size_t ws_size,
                              hipStream_t stream) {
    const float* q  = (const float*)d_in[0];
    const float* k  = (const float*)d_in[1];
    const float* v  = (const float*)d_in[2];
    const float* q_hop  = (const float*)d_in[3];
    const float* q_edge = (const float*)d_in[4];
    const float* k_hop  = (const float*)d_in[5];
    const float* k_edge = (const float*)d_in[6];
    const float* v_hop  = (const float*)d_in[7];
    const float* v_edge = (const float*)d_in[8];
    const int*   distance  = (const int*)d_in[9];
    const int*   edge_attr = (const int*)d_in[10];
    const float* Wq = (const float*)d_in[11]; const float* bq = (const float*)d_in[12];
    const float* Wk = (const float*)d_in[13]; const float* bk = (const float*)d_in[14];
    const float* Wv = (const float*)d_in[15]; const float* bv = (const float*)d_in[16];
    const float* Wo = (const float*)d_in[17]; const float* bo = (const float*)d_in[18];
    float* out = (float*)d_out;

    char* ws = (char*)d_ws;
    const size_t NBSD = (size_t)PB * PS * PD;     // 2,097,152
    const size_t NDD  = (size_t)PD * PD;          // 1,048,576
    size_t off = 0;
    unsigned short* q_bf  = (unsigned short*)(ws + off); off += NBSD * 2;
    unsigned short* k_bf  = (unsigned short*)(ws + off); off += NBSD * 2;
    unsigned short* v_bf  = (unsigned short*)(ws + off); off += NBSD * 2;
    unsigned short* Wq_bf = (unsigned short*)(ws + off); off += NDD * 2;
    unsigned short* Wk_bf = (unsigned short*)(ws + off); off += NDD * 2;
    unsigned short* Wv_bf = (unsigned short*)(ws + off); off += NDD * 2;
    unsigned short* Wo_bf = (unsigned short*)(ws + off); off += NDD * 2;
    unsigned short* qh_bf = (unsigned short*)(ws + off); off += NBSD * 2;   // [B,H,S,Dh]
    unsigned short* kh_bf = (unsigned short*)(ws + off); off += NBSD * 2;   // [B,H,S,Dh]
    unsigned short* vT_bf = (unsigned short*)(ws + off); off += NBSD * 2;   // [B,H,Dh,S]
    float* qhopT  = (float*)(ws + off); off += (size_t)PB * PH * PS * PT * 4;
    float* qedgeT = (float*)(ws + off); off += (size_t)PB * PH * PS * PE * 4;
    float* khopT  = (float*)(ws + off); off += (size_t)PB * PH * PS * PT * 4;
    float* kedgeT = (float*)(ws + off); off += (size_t)PB * PH * PS * PE * 4;
    unsigned short* attout_bf = (unsigned short*)(ws + off); off += NBSD * 2;
    (void)ws_size; (void)n_in; (void)in_sizes; (void)out_size;

    // 1) convert to bf16
    k_cvt_bf16<<<dim3((unsigned)(NBSD / 512)), 256, 0, stream>>>(q, q_bf, (int)NBSD);
    k_cvt_bf16<<<dim3((unsigned)(NBSD / 512)), 256, 0, stream>>>(k, k_bf, (int)NBSD);
    k_cvt_bf16<<<dim3((unsigned)(NBSD / 512)), 256, 0, stream>>>(v, v_bf, (int)NBSD);
    k_cvt_bf16<<<dim3((unsigned)(NDD  / 512)), 256, 0, stream>>>(Wq, Wq_bf, (int)NDD);
    k_cvt_bf16<<<dim3((unsigned)(NDD  / 512)), 256, 0, stream>>>(Wk, Wk_bf, (int)NDD);
    k_cvt_bf16<<<dim3((unsigned)(NDD  / 512)), 256, 0, stream>>>(Wv, Wv_bf, (int)NDD);
    k_cvt_bf16<<<dim3((unsigned)(NDD  / 512)), 256, 0, stream>>>(Wo, Wo_bf, (int)NDD);

    // 2) projections (M=B*S=2048, N=D=1024, K=D=1024), 128x64 block tiles
    dim3 gg(2048 / 128, 1024 / 64);
    k_gemm_bf16<<<gg, 256, 0, stream>>>(q_bf, Wq_bf, bq, nullptr, qh_bf, 2048, 1024, 1024, 0);
    k_gemm_bf16<<<gg, 256, 0, stream>>>(k_bf, Wk_bf, bk, nullptr, kh_bf, 2048, 1024, 1024, 0);
    k_gemm_bf16<<<gg, 256, 0, stream>>>(v_bf, Wv_bf, bv, nullptr, vT_bf, 2048, 1024, 1024, 2);

    // 3) hop/edge bias tables
    k_tables<<<dim3(PB * PH * PS), 64, 0, stream>>>(qh_bf, kh_bf, q_hop, q_edge, k_hop, k_edge,
                                                    qhopT, qedgeT, khopT, kedgeT);

    // 4) fused attention
    k_attn<<<dim3(PS / 32, PH, PB), 256, 0, stream>>>(qh_bf, kh_bf, vT_bf,
                                                      qhopT, qedgeT, khopT, kedgeT,
                                                      distance, edge_attr, v_hop, v_edge,
                                                      attout_bf);

    // 5) output projection -> d_out (f32)
    k_gemm_bf16<<<gg, 256, 0, stream>>>(attout_bf, Wo_bf, bo, out, nullptr, 2048, 1024, 1024, 1);
}
